// PointwiseInferenceLayer_63239098466586
// MI455X (gfx1250) — matrix operand order repaired and checked by reading, not compiled, not run on hardware
//
#include <hip/hip_runtime.h>
#include <hip/hip_bf16.h>
#include <math.h>

// ---------------------------------------------------------------------------
// Problem constants
// ---------------------------------------------------------------------------
#define B_   32
#define C_   512
#define P_   784          // H*W = 28*28
#define M_   200          // NC*NM
#define NC_  50
#define NM_  4
#define K_   5
#define OCC  0.3f

// Tiling for the main per-pixel GEMM kernel
#define MT     32         // m-rows per block (2 WMMA N-tiles)
#define PT     8          // pixels per block (1 wave per pixel, 8 waves)
#define CT     32         // K (channel) chunk per stage == one WMMA K step
#define NCHUNK (C_/CT)    // 16

typedef __attribute__((ext_vector_type(16))) __bf16 v16bf;
typedef __attribute__((ext_vector_type(8)))  float  v8f;
typedef __attribute__((__vector_size__(16))) int    v4i_t;

// ---------------------------------------------------------------------------
// CDNA5 async global->LDS copy (ASYNCcnt path), with portable fallback
// ---------------------------------------------------------------------------
#if defined(__has_builtin)
#if __has_builtin(__builtin_amdgcn_global_load_async_to_lds_b128)
#define ASYNC_CP 1
#endif
#endif

__device__ __forceinline__ float clamp01(float v) {
  return fminf(fmaxf(v, 0.0f), 1.0f);
}

__device__ __forceinline__ void cp16(const float* g, float* l) {
#ifdef ASYNC_CP
  __builtin_amdgcn_global_load_async_to_lds_b128(
      (__attribute__((address_space(1))) v4i_t*)(g),
      (__attribute__((address_space(3))) v4i_t*)(l),
      0, 0);
#else
  *(float4*)l = *(const float4*)g;
#endif
}

__device__ __forceinline__ void wait_cp() {
#ifdef ASYNC_CP
#if __has_builtin(__builtin_amdgcn_s_wait_asynccnt)
  __builtin_amdgcn_s_wait_asynccnt(0);
#else
  asm volatile("s_wait_asynccnt 0" ::: "memory");
#endif
#endif
}

// ---------------------------------------------------------------------------
// Kernel 0: zero per_model accumulator in workspace
// ---------------------------------------------------------------------------
__global__ __launch_bounds__(256) void pw_zero_kernel(float* p, int n) {
  int t = blockIdx.x * 256 + threadIdx.x;
  if (t < n) p[t] = 0.0f;
}

// ---------------------------------------------------------------------------
// Kernel 1: background log-likelihood  bg_log[b][p][k]
//   cl = L1-normalize(clamp(clutter,0,1)) over C; normalization folded into a
//   single scale of the dot product.
// ---------------------------------------------------------------------------
__global__ __launch_bounds__(256) void pw_bg_kernel(
    const float* __restrict__ x, const float* __restrict__ clutter,
    float* __restrict__ bgws) {
  __shared__ float scl[K_ * C_];
  __shared__ float sinv[K_];
  const int tid = threadIdx.x;

  for (int i = tid; i < K_ * C_; i += 256) scl[i] = clamp01(clutter[i]);
  __syncthreads();
  if (tid < K_) {
    float s = 0.0f;
    for (int c = 0; c < C_; ++c) s += scl[tid * C_ + c];
    sinv[tid] = 1.0f / fmaxf(s, 1e-12f);
  }
  __syncthreads();

  const int t = blockIdx.x * 256 + tid;
  if (t >= B_ * P_) return;
  const int b = t / P_;
  const int p = t - b * P_;
  float acc[K_] = {0.f, 0.f, 0.f, 0.f, 0.f};
  const float* xp = x + (size_t)b * C_ * P_ + p;
  for (int c = 0; c < C_; ++c) {
    float xv = xp[(size_t)c * P_];     // lanes have consecutive p -> coalesced
#pragma unroll
    for (int k = 0; k < K_; ++k) acc[k] += xv * scl[k * C_ + c];
  }
#pragma unroll
  for (int k = 0; k < K_; ++k)
    bgws[(size_t)(b * P_ + p) * K_ + k] = logf(acc[k] * sinv[k] * OCC + 1e-10f);
}

// ---------------------------------------------------------------------------
// Kernel 2: main per-pixel GEMM (WMMA bf16), fused normalize/log/max/sum
//   grid = (P_/PT = 98, ceil(M_/MT) = 7), block = 256 (8 waves, 1 wave/pixel)
// ---------------------------------------------------------------------------
__device__ __forceinline__ void issue_chunk(
    const float* __restrict__ x, const float* __restrict__ mix,
    float* smx, float* smw, int buf, int ch, int pBase, int mBase, int tid) {
  const int cBase = ch * CT;
  float* dx = smx + buf * (CT * B_ * PT);
  float* dw = smw + buf * (CT * MT * PT);
  // x tile: [CT][B_][PT] ; every (c,b) row = 8 contiguous pixels = 2 x 16B
  for (int i = tid; i < CT * B_ * 2; i += 256) {
    int h4 = (i & 1) * 4;
    int b  = (i >> 1) % B_;
    int c  = i / (B_ * 2);
    const float* g = x + ((size_t)b * C_ + cBase + c) * P_ + pBase + h4;
    cp16(g, dx + ((c * B_ + b) * PT + h4));
  }
  // mix tile: [CT][MT][PT]
  for (int i = tid; i < CT * MT * 2; i += 256) {
    int h4 = (i & 1) * 4;
    int m  = (i >> 1) % MT;
    int c  = i / (MT * 2);
    int mg = mBase + m; if (mg >= M_) mg = M_ - 1;   // clamp, masked later
    const float* g = mix + ((size_t)mg * C_ + cBase + c) * P_ + pBase + h4;
    cp16(g, dw + ((c * MT + m) * PT + h4));
  }
}

__global__ __launch_bounds__(256) void pw_fg_kernel(
    const float* __restrict__ x, const float* __restrict__ mix,
    const float* __restrict__ bgws, float* __restrict__ per_model) {
  __shared__ float smx[2 * CT * B_ * PT];     // 64 KB  (f32 staged, double buf)
  __shared__ float smw[2 * CT * MT * PT];     // 64 KB
  __shared__ float denoms[PT * MT];           // 1 KB   sum_c clamp(mix)
  __shared__ float smbg[PT * B_ * K_];        // 5 KB   bg_log for this p-chunk
  __shared__ float spart[MT * B_ * K_];       // 20 KB  per-block per_model part

  const int tid   = threadIdx.x;
  const int lane  = tid & 31;
  const int wv    = tid >> 5;                 // wave id = local pixel
  const int pBase = blockIdx.x * PT;
  const int mBase = blockIdx.y * MT;

  // ---- init shared state ----
  denoms[tid] = 0.0f;                          // PT*MT == 256
  for (int i = tid; i < MT * B_ * K_; i += 256) spart[i] = 0.0f;
  {
    int b = tid >> 3, p = tid & 7;             // 256 == B_*PT pairs
#pragma unroll
    for (int k = 0; k < K_; ++k)
      smbg[(p * B_ + b) * K_ + k] = bgws[((size_t)b * P_ + pBase + p) * K_ + k];
  }

  v8f acc[2][2] = {};                          // [mtile j][btile t]

  issue_chunk(x, mix, smx, smw, 0, 0, pBase, mBase, tid);
  wait_cp();
  __syncthreads();

  const int p  = wv;
  const int hi = lane >> 4;
  const int lo = lane & 15;

  for (int ch = 0; ch < NCHUNK; ++ch) {
    const int buf = ch & 1;
    if (ch + 1 < NCHUNK)
      issue_chunk(x, mix, smx, smw, buf ^ 1, ch + 1, pBase, mBase, tid);

    const float* sx = smx + buf * (CT * B_ * PT);
    const float* sw = smw + buf * (CT * MT * PT);

    // denominator accumulation: one (p,m) pair per thread
    {
      int dp = tid >> 5, dm = tid & 31;
      float s = 0.0f;
#pragma unroll
      for (int c = 0; c < CT; ++c)
        s += clamp01(sw[(c * MT + dm) * PT + dp]);
      denoms[dp * MT + dm] += s;
    }

    // A fragments (x): 16x32 bf16 per ISA layout — lane<16: K {0..7,16..23},
    // lane>=16: K {8..15,24..31}; row M = lane&15
    v16bf A[2];
#pragma unroll
    for (int t = 0; t < 2; ++t) {
      int b = t * 16 + lo;
#pragma unroll
      for (int e = 0; e < 8; ++e) {
        A[t][e]     = (__bf16)sx[((hi * 8 + e) * B_ + b) * PT + p];
        A[t][e + 8] = (__bf16)sx[((16 + hi * 8 + e) * B_ + b) * PT + p];
      }
    }
    // B fragments (mix, clamped): 32x16 bf16 — lane<16: K {0..7,16..23}, lane>=16: K {8..15,24..31}
#pragma unroll
    for (int j = 0; j < 2; ++j) {
      int m = j * 16 + lo;
      v16bf Bf;
#pragma unroll
      for (int e = 0; e < 16; ++e)
        Bf[e] = (__bf16)clamp01(sw[(((e >> 3) * 16 + hi * 8 + (e & 7)) * MT + m) * PT + p]);
      acc[j][0] = __builtin_amdgcn_wmma_f32_16x16x32_bf16(
          false, A[0], false, Bf, (short)0, acc[j][0], false, false);
      acc[j][1] = __builtin_amdgcn_wmma_f32_16x16x32_bf16(
          false, A[1], false, Bf, (short)0, acc[j][1], false, false);
    }

    wait_cp();
    __syncthreads();
  }

  // ---- epilogue: normalize, log, max with bg, reduce over pixels in LDS ----
#pragma unroll
  for (int j = 0; j < 2; ++j) {
    int ml = j * 16 + lo;
    int mg = mBase + ml;
    float inv = 1.0f / fmaxf(denoms[p * MT + ml], 1e-12f);
    if (mg < M_) {
#pragma unroll
      for (int t = 0; t < 2; ++t) {
#pragma unroll
        for (int r = 0; r < 8; ++r) {
          int b = t * 16 + hi * 8 + r;     // D layout: VGPR r -> M=r / r+8
          float fg = logf(acc[j][t][r] * inv * (1.0f - OCC) + 1e-10f);
#pragma unroll
          for (int k = 0; k < K_; ++k)
            atomicAdd(&spart[(ml * B_ + b) * K_ + k],
                      fmaxf(fg, smbg[(p * B_ + b) * K_ + k]));
        }
      }
    }
  }
  __syncthreads();

  // flush block partials to global per_model
  for (int i = tid; i < MT * B_ * K_; i += 256) {
    int ml  = i / (B_ * K_);
    int rem = i - ml * (B_ * K_);
    int b   = rem / K_;
    int k   = rem - b * K_;
    int mg  = mBase + ml;
    if (mg < M_)
      atomicAdd(&per_model[((size_t)b * M_ + mg) * K_ + k], spart[i]);
  }
}

// ---------------------------------------------------------------------------
// Kernel 3: out[b][nc] = max over nm,k of per_model[b][nc*4+nm][k]
// ---------------------------------------------------------------------------
__global__ __launch_bounds__(256) void pw_reduce_kernel(
    const float* __restrict__ pm, float* __restrict__ out) {
  int t = blockIdx.x * 256 + threadIdx.x;
  if (t >= B_ * NC_) return;
  int b = t / NC_, nc = t - b * NC_;
  float best = -INFINITY;
#pragma unroll
  for (int nm = 0; nm < NM_; ++nm)
#pragma unroll
    for (int k = 0; k < K_; ++k)
      best = fmaxf(best, pm[((size_t)b * M_ + nc * NM_ + nm) * K_ + k]);
  out[t] = best;
}

// ---------------------------------------------------------------------------
// Launch
// ---------------------------------------------------------------------------
extern "C" void kernel_launch(void* const* d_in, const int* in_sizes, int n_in,
                              void* d_out, int out_size, void* d_ws, size_t ws_size,
                              hipStream_t stream) {
  const float* x       = (const float*)d_in[0];   // [32,512,28,28]
  const float* mix     = (const float*)d_in[1];   // [200,512,28,28]
  const float* clutter = (const float*)d_in[2];   // [5,512]
  float* out = (float*)d_out;                     // [32,50]

  float* per_model = (float*)d_ws;                // [32][200][5] = 32000 f32
  float* bgws      = per_model + (size_t)B_ * M_ * K_;  // [32][784][5]

  pw_zero_kernel<<<(B_ * M_ * K_ + 255) / 256, 256, 0, stream>>>(
      per_model, B_ * M_ * K_);
  pw_bg_kernel<<<(B_ * P_) / 256, 256, 0, stream>>>(x, clutter, bgws);
  pw_fg_kernel<<<dim3(P_ / PT, (M_ + MT - 1) / MT), 256, 0, stream>>>(
      x, mix, bgws, per_model);
  pw_reduce_kernel<<<(B_ * NC_ + 255) / 256, 256, 0, stream>>>(per_model, out);
}
